// GCN_drop_1597727834314
// MI455X (gfx1250) — matrix operand order repaired and checked by reading, not compiled, not run on hardware
//
#include <hip/hip_runtime.h>

// ---------------------------------------------------------------------------
// Types for CDNA5 WMMA (wave32)
// ---------------------------------------------------------------------------
typedef __attribute__((ext_vector_type(16))) __bf16         v16bf;
typedef __attribute__((ext_vector_type(16))) unsigned short v16u16;
typedef __attribute__((ext_vector_type(8)))  unsigned short v8u16;
typedef __attribute__((ext_vector_type(8)))  float          v8f;

static __device__ __forceinline__ unsigned short f2bf(float f) {
    union { float f; unsigned int u; } c; c.f = f;
    unsigned int u = c.u;
    unsigned int r = u + 0x7FFFu + ((u >> 16) & 1u);   // round-to-nearest-even
    return (unsigned short)(r >> 16);
}

// ---------------------------------------------------------------------------
// Degree accumulation (fp32 atomics; arrays are L2-resident)
// ---------------------------------------------------------------------------
__global__ void degree_kernel(const int* __restrict__ src, const int* __restrict__ dst,
                              float* __restrict__ outdeg, float* __restrict__ indeg, int E) {
    int e = blockIdx.x * blockDim.x + threadIdx.x;
    if (e < E) {
        unsafeAtomicAdd(&outdeg[src[e]], 1.0f);
        unsafeAtomicAdd(&indeg[dst[e]],  1.0f);
    }
}

__global__ void norm_kernel(float* __restrict__ outn, float* __restrict__ inn, int N) {
    int i = blockIdx.x * blockDim.x + threadIdx.x;
    if (i < N) {
        outn[i] = __frsqrt_rn(fmaxf(outn[i], 1.0f));
        inn[i]  = __frsqrt_rn(fmaxf(inn[i],  1.0f));
    }
}

// ---------------------------------------------------------------------------
// Pre-convert A = X * rowScale[m] to bf16, row-major (bandwidth-bound pass).
// 8 elements per thread -> float4 x2 in, v8u16 (16B) out.
// ---------------------------------------------------------------------------
template<int K>
__global__ void convert_scale_kernel(const float* __restrict__ X, const float* __restrict__ rs,
                                     unsigned short* __restrict__ out, long long nvec) {
    long long v = (long long)blockIdx.x * blockDim.x + threadIdx.x;
    if (v >= nvec) return;
    long long base = v << 3;                 // 8 floats per thread
    int n = (int)(base / K);                 // K is power of two -> shift
    float s = rs[n];
    const float4 a0 = *(const float4*)(X + base);
    const float4 a1 = *(const float4*)(X + base + 4);
    v8u16 o;
    o[0] = f2bf(a0.x * s); o[1] = f2bf(a0.y * s);
    o[2] = f2bf(a0.z * s); o[3] = f2bf(a0.w * s);
    o[4] = f2bf(a1.x * s); o[5] = f2bf(a1.y * s);
    o[6] = f2bf(a1.z * s); o[7] = f2bf(a1.w * s);
    *(v8u16*)(out + base) = o;
}

// ---------------------------------------------------------------------------
// Repack W (KxN, f32 row-major) into bf16 B-fragments:
//   P[((kt*NT + nt)*32 + lane)*16 + e] = bf16( W[(kt*32 + (lane>>4)*16 + e)*N + nt*16 + (lane&15)] )
// Contiguous k order (same as the A loads, not the ISA table order): lanes 0-15 hold K=0-15, lanes 16-31
// hold K=16-31, element e = K within half (2 packed per dword).
// ---------------------------------------------------------------------------
__global__ void pack_w_kernel(const float* __restrict__ W, unsigned short* __restrict__ P,
                              int K, int N) {
    int idx = blockIdx.x * blockDim.x + threadIdx.x;
    if (idx >= K * N) return;
    int e    = idx & 15;
    int lane = (idx >> 4) & 31;
    int t    = idx >> 9;                 // kt*NT + nt
    int NT   = N >> 4;
    int kt   = t / NT;
    int nt   = t - kt * NT;
    int n    = nt * 16 + (lane & 15);
    int k    = kt * 32 + (lane >> 4) * 16 + e;
    P[idx] = f2bf(W[(size_t)k * N + n]);
}

// ---------------------------------------------------------------------------
// WMMA GEMM: C[M,N] = A_bf16[M,K] @ Wpacked (f32 accumulate).
// One wave per 16x16 tile; blockDim = 32*(N/16): all waves of a block share
// the same A-tile through the WGP cache. Two accumulator chains break the
// WMMA->WMMA RAW dependency.
// A fragment (16-bit 16x32 layout): m = lane&15, half = lane>>4,
//   e<8  -> K = kk + half*16 + e        (8 contiguous bf16 = one b128 load)
//   e>=8 -> K = kk + half*16 + e        (8 contiguous bf16 = one b128 load)
// ---------------------------------------------------------------------------
template<int K, int N>
__global__ __launch_bounds__(32 * (N / 16))
void gemm_bf16(const unsigned short* __restrict__ A, const unsigned short* __restrict__ Bp,
               float* __restrict__ C, int M) {
    constexpr int NT = N / 16;
    const int wave  = threadIdx.x >> 5;      // = ntile
    const int lane  = threadIdx.x & 31;
    const int half  = lane >> 4;
    const int mrow  = lane & 15;
    const int mtile = blockIdx.x;

    int m = mtile * 16 + mrow;
    if (m >= M) m = M - 1;                   // clamp loads; stores guarded below
    const unsigned short* __restrict__ ar = A + (size_t)m * K;
    const unsigned short* __restrict__ bp = Bp + (((size_t)wave * 32 + lane) << 4);
    constexpr size_t BSTEP = (size_t)NT * 32 * 16;   // elements per K-step in packed B

    v8f acc0 = {}, acc1 = {};
#pragma unroll
    for (int kk = 0; kk < K; kk += 64) {
        {
            v8u16 lo = *(const v8u16*)(ar + kk + half * 16);
            v8u16 hi = *(const v8u16*)(ar + kk + 8 + half * 16);
            v16bf a = __builtin_bit_cast(v16bf, __builtin_shufflevector(
                lo, hi, 0, 1, 2, 3, 4, 5, 6, 7, 8, 9, 10, 11, 12, 13, 14, 15));
            v16bf b = __builtin_bit_cast(v16bf,
                *(const v16u16*)(bp + (size_t)(kk >> 5) * BSTEP));
            acc0 = __builtin_amdgcn_wmma_f32_16x16x32_bf16(false, a, false, b,
                                                           (short)0, acc0, false, false);
        }
        {
            v8u16 lo = *(const v8u16*)(ar + kk + 32 + half * 16);
            v8u16 hi = *(const v8u16*)(ar + kk + 40 + half * 16);
            v16bf a = __builtin_bit_cast(v16bf, __builtin_shufflevector(
                lo, hi, 0, 1, 2, 3, 4, 5, 6, 7, 8, 9, 10, 11, 12, 13, 14, 15));
            v16bf b = __builtin_bit_cast(v16bf,
                *(const v16u16*)(bp + (size_t)((kk >> 5) + 1) * BSTEP));
            acc1 = __builtin_amdgcn_wmma_f32_16x16x32_bf16(false, a, false, b,
                                                           (short)0, acc1, false, false);
        }
    }

    // D layout: VGPR r holds row (half*8 + r), col = lane&15
    const int col = wave * 16 + mrow;
    if (mtile * 16 + 16 <= M) {              // uniform fast path (M % 16 == 0 here)
        float* c = C + (size_t)(mtile * 16 + half * 8) * N + col;
#pragma unroll
        for (int r = 0; r < 8; ++r) c[(size_t)r * N] = acc0[r] + acc1[r];
    } else {
#pragma unroll
        for (int r = 0; r < 8; ++r) {
            int row = mtile * 16 + half * 8 + r;
            if (row < M) C[(size_t)row * N + col] = acc0[r] + acc1[r];
        }
    }
}

// ---------------------------------------------------------------------------
// Edge gather + scatter-add: thread per (edge, feature) -> coalesced over F
// ---------------------------------------------------------------------------
template<int F>
__global__ void scatter_add_kernel(const float* __restrict__ H, const int* __restrict__ src,
                                   const int* __restrict__ dst, float* __restrict__ agg,
                                   long long total) {
    long long idx = (long long)blockIdx.x * blockDim.x + threadIdx.x;
    if (idx >= total) return;
    int e = (int)(idx >> (F == 128 ? 7 : 6));
    int f = (int)(idx & (F - 1));
    int s = src[e], d = dst[e];
    unsafeAtomicAdd(&agg[(size_t)d * F + f], H[(size_t)s * F + f]);
}

// ---------------------------------------------------------------------------
// Layer-1 epilogue: Tb = bf16( relu(agg * in_norm + b1) * out_norm )
// (produces the pre-scaled bf16 input for GEMM2)
// ---------------------------------------------------------------------------
template<int F>
__global__ void epilogue_relu_bf16_kernel(const float* __restrict__ agg,
                                          const float* __restrict__ inn,
                                          const float* __restrict__ outn,
                                          const float* __restrict__ bias,
                                          unsigned short* __restrict__ out, long long total) {
    long long idx = (long long)blockIdx.x * blockDim.x + threadIdx.x;
    if (idx >= total) return;
    int n = (int)(idx >> 7);                 // F == 128
    int f = (int)(idx & (F - 1));
    float v = fmaxf(agg[idx] * inn[n] + bias[f], 0.0f);
    out[idx] = f2bf(v * outn[n]);
}

// ---------------------------------------------------------------------------
// Final epilogue: out = agg * in_norm + b2  (f32)
// ---------------------------------------------------------------------------
template<int F>
__global__ void epilogue_f32_kernel(const float* __restrict__ agg, const float* __restrict__ inn,
                                    const float* __restrict__ bias, float* __restrict__ out,
                                    long long total) {
    long long idx = (long long)blockIdx.x * blockDim.x + threadIdx.x;
    if (idx >= total) return;
    int n = (int)(idx >> 6);                 // F == 64
    int f = (int)(idx & (F - 1));
    out[idx] = agg[idx] * inn[n] + bias[f];
}

// ---------------------------------------------------------------------------
// Launch
// ---------------------------------------------------------------------------
extern "C" void kernel_launch(void* const* d_in, const int* in_sizes, int n_in,
                              void* d_out, int out_size, void* d_ws, size_t ws_size,
                              hipStream_t stream) {
    const float* feat = (const float*)d_in[0];
    const int*   esrc = (const int*)  d_in[1];
    const int*   edst = (const int*)  d_in[2];
    const float* W1   = (const float*)d_in[3];
    const float* b1   = (const float*)d_in[4];
    const float* W2   = (const float*)d_in[5];
    const float* b2   = (const float*)d_in[6];

    constexpr int FIN = 512, FH = 128, FO = 64;
    const int E = in_sizes[1];
    const int M = in_sizes[0] / FIN;   // 50000

    // Workspace layout (256B aligned); AGG aliases Xb (dead after GEMM1).
    char* ws = (char*)d_ws;
    size_t off = 0;
    auto alloc = [&](size_t bytes) { size_t o = off; off = (off + bytes + 255) & ~(size_t)255; return o; };
    float*          outn = (float*)         (ws + alloc((size_t)M * 4));
    float*          inn  = (float*)         (ws + alloc((size_t)M * 4));
    unsigned short* W1p  = (unsigned short*)(ws + alloc((size_t)FIN * FH * 2));
    unsigned short* W2p  = (unsigned short*)(ws + alloc((size_t)FH * FO * 2));
    unsigned short* Xb   = (unsigned short*)(ws + alloc((size_t)M * FIN * 2)); // bf16 X*outn
    float*          AGG  = (float*)Xb;                                          // alias (needs M*FH*4 <= M*FIN*2)
    float*          T1   = (float*)         (ws + alloc((size_t)M * FH * 4));  // GEMM1 out (f32 messages)
    unsigned short* Tb   = (unsigned short*)(ws + alloc((size_t)M * FH * 2));  // bf16 relu(..)*outn
    float*          H2   = (float*)         (ws + alloc((size_t)M * FO * 4));  // GEMM2 out (f32 messages)
    (void)ws_size; (void)n_in; (void)out_size;

    const int mtiles = (M + 15) / 16;

    // 1) degrees -> norms
    hipMemsetAsync(outn, 0, (size_t)M * 4, stream);
    hipMemsetAsync(inn,  0, (size_t)M * 4, stream);
    degree_kernel<<<(E + 255) / 256, 256, 0, stream>>>(esrc, edst, outn, inn, E);
    norm_kernel<<<(M + 255) / 256, 256, 0, stream>>>(outn, inn, M);

    // 2) pack weights; pre-scale + convert X to bf16
    pack_w_kernel<<<(FIN * FH + 255) / 256, 256, 0, stream>>>(W1, W1p, FIN, FH);
    pack_w_kernel<<<(FH * FO + 255) / 256, 256, 0, stream>>>(W2, W2p, FH, FO);
    {
        long long nvec = (long long)M * FIN / 8;
        convert_scale_kernel<FIN><<<(int)((nvec + 255) / 256), 256, 0, stream>>>(feat, outn, Xb, nvec);
    }

    // 3) layer 1 GEMM: T1 = Xb @ W1  (bf16 WMMA, f32 acc)
    gemm_bf16<FIN, FH><<<mtiles, 32 * (FH / 16), 0, stream>>>(Xb, W1p, T1, M);

    // 4) edge aggregation (f32, L2-resident atomics)  -- AGG aliases Xb (now dead)
    hipMemsetAsync(AGG, 0, (size_t)M * FH * 4, stream);
    {
        long long total = (long long)E * FH;
        scatter_add_kernel<FH><<<(int)((total + 255) / 256), 256, 0, stream>>>(T1, esrc, edst, AGG, total);
    }
    // 5) Tb = bf16( relu(agg*inn + b1) * outn )
    {
        long long total = (long long)M * FH;
        epilogue_relu_bf16_kernel<FH><<<(int)((total + 255) / 256), 256, 0, stream>>>(AGG, inn, outn, b1, Tb, total);
    }

    // 6) layer 2 GEMM: H2 = Tb @ W2
    gemm_bf16<FH, FO><<<mtiles, 32 * (FO / 16), 0, stream>>>(Tb, W2p, H2, M);

    // 7) edge aggregation layer 2 (reuse AGG region)
    hipMemsetAsync(AGG, 0, (size_t)M * FO * 4, stream);
    {
        long long total = (long long)E * FO;
        scatter_add_kernel<FO><<<(int)((total + 255) / 256), 256, 0, stream>>>(H2, esrc, edst, AGG, total);
    }
    // 8) out = agg * in_norm + b2
    {
        long long total = (long long)M * FO;
        epilogue_f32_kernel<FO><<<(int)((total + 255) / 256), 256, 0, stream>>>(AGG, inn, b2, (float*)d_out, total);
    }
}
